// CategoricalLstmDecoder_12927851561093
// MI455X (gfx1250) — matrix operand order repaired and checked by reading, not compiled, not run on hardware
//
#include <hip/hip_runtime.h>
#include <hip/hip_bf16.h>

typedef __attribute__((ext_vector_type(16))) _Float16 v16h;
typedef __attribute__((ext_vector_type(8)))  _Float16 v8h;
typedef __attribute__((ext_vector_type(8)))  float    v8f;

#define BATCH    64
#define HID      1024
#define VOCAB    512
#define SEQ      512
#define NWG      32
#define NTHREADS 256
#define HPW      32            // hidden units owned per workgroup (per layer)
#define ASTR     2056          // LDS A-buffer row stride (f16), padded vs 2048
#define GSTR     132           // LDS gates row stride (f32), padded vs 128

#define KT0 48                 // (512 + 1024) / 32 k-tiles, layer 0
#define KT1 64                 // (1024 + 1024) / 32 k-tiles, layer 1
#define KTO 32                 // 1024 / 32 k-tiles, output projection

// ---- workspace layout (bytes) ----
#define WS_BAR 0
#define WS_B0  256
#define WS_B1  (WS_B0 + 4096*4)
#define WS_H0  (WS_B1 + 4096*4)
#define WS_H1  (WS_H0 + 2*BATCH*HID*2)
#define WS_LOG (WS_H1 + 2*BATCH*HID*2)
#define WS_BP0 (WS_LOG + BATCH*VOCAB*4)
#define WS_BP1 (WS_BP0 + 256*KT0*512*2)
#define WS_BPO (WS_BP1 + 256*KT1*512*2)

__device__ __forceinline__ float sigf(float x) { return 1.0f / (1.0f + __expf(-x)); }

// ------------------------------------------------------------------
// Pack B matrices (gates = A[64xK] * B[KxN]) into per-lane v16h WMMA
// fragments.  B[k][n] = W[n*K + k] (W is [N,K] row-major; gates = x @ W^T).
// Fragment layout (V_WMMA_F32_16X16X32_F16):
//   lane l<16 : column N = nt*16+l,      K = kt*32 + {0..7, 16..23}  (elements 0..15)
//   lane l>=16: column N = nt*16+(l-16), K = kt*32 + {8..15, 24..31}
// Flat index: (((gnt*KT + kt)*32 + lane)*16 + e)
// ------------------------------------------------------------------
__global__ void pack_b_kernel(const float* __restrict__ wih, const float* __restrict__ whh,
                              int Kx, int Kh, int KT, long total, _Float16* __restrict__ out) {
  long i = (long)blockIdx.x * blockDim.x + threadIdx.x;
  if (i >= total) return;
  int  e    = (int)(i & 15);
  int  lane = (int)((i >> 4) & 31);
  long r    = i >> 9;
  int  kt   = (int)(r % KT);
  int  gnt  = (int)(r / KT);
  int  N    = gnt * 16 + (lane & 15);
  int  K    = kt * 32 + ((lane >> 4) << 3) + (e & 7) + ((e >> 3) << 4);
  float v = (K < Kx) ? wih[(long)N * Kx + K] : whh[(long)N * Kh + (K - Kx)];
  out[i] = (_Float16)v;
}

// h_init = tanh(z @ fc_w^T + fc_b); also fuse biases and reset barrier.
__global__ void init_kernel(const float* __restrict__ z, const float* __restrict__ fcw,
                            const float* __restrict__ fcb,
                            const float* __restrict__ bih0, const float* __restrict__ bhh0,
                            const float* __restrict__ bih1, const float* __restrict__ bhh1,
                            _Float16* __restrict__ h0, _Float16* __restrict__ h1,
                            float* __restrict__ b0, float* __restrict__ b1,
                            unsigned* __restrict__ bar) {
  int i = blockIdx.x * blockDim.x + threadIdx.x;   // 0..65535
  int b = i >> 10, u = i & 1023;
  float acc = fcb[u];
  for (int k = 0; k < 256; ++k) acc += z[b * 256 + k] * fcw[u * 256 + k];
  _Float16 h = (_Float16)tanhf(acc);
  h0[i] = h;                 // buffer 0 (read at t=0)
  h1[i] = h;
  if (i < 4096) { b0[i] = bih0[i] + bhh0[i]; b1[i] = bih1[i] + bhh1[i]; }
  if (i < 2)    { bar[i] = 0u; }
}

// ------------------------------------------------------------------
// Grid-wide sense-reversing barrier (persistent kernel, 32 WGs).
// ------------------------------------------------------------------
__device__ __forceinline__ void grid_barrier(unsigned* cnt, unsigned* gen) {
  __syncthreads();
  if (threadIdx.x == 0) {
    __threadfence();
    unsigned g = __atomic_load_n(gen, __ATOMIC_ACQUIRE);
    if (atomicAdd(cnt, 1u) == NWG - 1u) {
      atomicExch(cnt, 0u);
      __threadfence();
      atomicAdd(gen, 1u);
    } else {
      while (__atomic_load_n(gen, __ATOMIC_ACQUIRE) == g)
        __builtin_amdgcn_s_sleep(2);
    }
    __threadfence();
  }
  __syncthreads();
}

// Stage h ([64][1024] f16 row-major, global) into LDS A-buffer (stride ASTR).
__device__ __forceinline__ void stage_h(const _Float16* __restrict__ src, _Float16* dst, int tid) {
  for (int i = tid; i < BATCH * (HID / 8); i += NTHREADS) {
    int r = i >> 7, cc = i & 127;
    *(v8h*)(dst + r * ASTR + cc * 8) = *(const v8h*)(src + (long)r * HID + cc * 8);
  }
}

// 4 m-tiles x 1 n-tile GEMM over KT k-tiles.  A from LDS (row-major, ASTR),
// B from pre-packed global fragments.
__device__ __forceinline__ void gemm_tile4(const _Float16* __restrict__ bpack, long bBase,
                                           const _Float16* aLds, int KT, int lane, v8f acc[4]) {
  const v16h* bp   = ((const v16h*)bpack) + bBase + lane;
  const int   mrow = lane & 15;
  const int   klo  = (lane & 16) ? 8 : 0;
  for (int kt = 0; kt < KT; ++kt) {
    if (kt + 8 < KT) __builtin_prefetch(&bp[(long)(kt + 8) * 32], 0, 1);
    v16h bfrag = bp[(long)kt * 32];
    const _Float16* abase = aLds + kt * 32 + klo;
#pragma unroll
    for (int mt = 0; mt < 4; ++mt) {
      const _Float16* ap = abase + (mt * 16 + mrow) * ASTR;
      v8h lo = *(const v8h*)ap;         // K-run [klo, klo+8)
      v8h hi = *(const v8h*)(ap + 16);  // K-run [klo+16, klo+24)
      v16h af = __builtin_shufflevector(lo, hi, 0,1,2,3,4,5,6,7,8,9,10,11,12,13,14,15);
      acc[mt] = __builtin_amdgcn_wmma_f32_16x16x32_f16(false, af, false, bfrag,
                                                       (short)0, acc[mt], false, false);
    }
  }
}

// One LSTM layer slice: gates GEMM -> exchange via LDS -> cell update.
__device__ __forceinline__ void gate_layer(const _Float16* __restrict__ bpack,
                                           const float* __restrict__ bias,
                                           const _Float16* aLds, float* gLds, float* cSt,
                                           _Float16* __restrict__ hOut,
                                           int KT, int wg, int wid, int lane, int tid) {
  const int gate = wid >> 1;
  const int gnt  = gate * 64 + wg * 2 + (wid & 1);   // == global_gate_col / 16
  v8f acc[4] = {{}, {}, {}, {}};
  gemm_tile4(bpack, (long)gnt * KT * 32, aLds, KT, lane, acc);

  // epilogue: C/D layout -> gates LDS, +bias
  const int nloc  = wid * 16 + (lane & 15);
  const int uu    = nloc & 31;
  const float bj  = bias[gate * HID + wg * HPW + uu];
  const int rbase = (lane & 16) ? 8 : 0;
#pragma unroll
  for (int mt = 0; mt < 4; ++mt)
#pragma unroll
    for (int r = 0; r < 8; ++r)
      gLds[(mt * 16 + rbase + r) * GSTR + nloc] = acc[mt][r] + bj;
  __syncthreads();

  // cell update for this WG's 32 hidden units x 64 batch rows
  for (int i = tid; i < BATCH * HPW; i += NTHREADS) {
    int b = i >> 5, u = i & 31;
    float gi = gLds[b * GSTR + u];
    float gf = gLds[b * GSTR + 32 + u];
    float gg = gLds[b * GSTR + 64 + u];
    float go = gLds[b * GSTR + 96 + u];
    float c  = cSt[i];
    float cn = sigf(gf) * c + sigf(gi) * tanhf(gg);
    cSt[i] = cn;
    hOut[b * HID + wg * HPW + u] = (_Float16)(sigf(go) * tanhf(cn));
  }
}

__global__ void __launch_bounds__(NTHREADS, 1)
lstm_main(const _Float16* __restrict__ bp0, const _Float16* __restrict__ bp1,
          const _Float16* __restrict__ bpo,
          const float* __restrict__ b0, const float* __restrict__ b1,
          const float* __restrict__ out_b,
          _Float16* __restrict__ h0buf, _Float16* __restrict__ h1buf,
          float* __restrict__ logits, float* __restrict__ out, unsigned* bar) {
  extern __shared__ char smem[];
  _Float16* aLds = (_Float16*)smem;                                   // 64 x 2056 f16
  float*    gLds = (float*)(smem + BATCH * ASTR * 2);                 // 64 x 132 f32
  float*    cSt  = (float*)(smem + BATCH * ASTR * 2 + BATCH * GSTR * 4); // [2][64*32] f32

  const int tid  = threadIdx.x;
  const int lane = tid & 31;
  const int wid  = tid >> 5;
  const int wg   = blockIdx.x;

  for (int i = tid; i < 2 * BATCH * HPW; i += NTHREADS) cSt[i] = 0.0f;

  unsigned* cnt = bar;
  unsigned* gen = bar + 1;

  for (int t = 0; t < SEQ; ++t) {
    const int p = t & 1;
    const _Float16* h0r = h0buf + p * (BATCH * HID);
    _Float16*       h0w = h0buf + (1 - p) * (BATCH * HID);
    const _Float16* h1r = h1buf + p * (BATCH * HID);
    _Float16*       h1w = h1buf + (1 - p) * (BATCH * HID);

    // ---------------- Phase A: x = softmax(prev logits); layer 0 ----------------
    if (t == 0) {
      v8h zz = {};
      for (int i = tid; i < BATCH * (VOCAB / 8); i += NTHREADS) {
        int r = i >> 6, cc = i & 63;
        *(v8h*)(aLds + r * ASTR + cc * 8) = zz;
      }
    } else {
      for (int r = wid; r < BATCH; r += 8) {            // 8 rows per wave
        const float* lr = logits + r * VOCAB;
        float vals[16];
        float mx = -3.4e38f;
#pragma unroll
        for (int j = 0; j < 16; ++j) { vals[j] = lr[lane + j * 32]; mx = fmaxf(mx, vals[j]); }
        for (int o = 16; o; o >>= 1) mx = fmaxf(mx, __shfl_xor(mx, o, 32));
        float s = 0.0f;
#pragma unroll
        for (int j = 0; j < 16; ++j) { vals[j] = __expf(vals[j] - mx); s += vals[j]; }
        for (int o = 16; o; o >>= 1) s += __shfl_xor(s, o, 32);
        float inv = 1.0f / s;
#pragma unroll
        for (int j = 0; j < 16; ++j) aLds[r * ASTR + lane + j * 32] = (_Float16)(vals[j] * inv);
      }
    }
    stage_h(h0r, aLds + VOCAB, tid);                    // A = [x(512) || h0(1024)]
    __syncthreads();
    gate_layer(bp0, b0, aLds, gLds, cSt, h0w, KT0, wg, wid, lane, tid);
    grid_barrier(cnt, gen);

    // ---------------- Phase B: layer 1 ----------------
    stage_h(h0w, aLds, tid);                            // A = [h0_new || h1_prev]
    stage_h(h1r, aLds + HID, tid);
    __syncthreads();
    gate_layer(bp1, b1, aLds, gLds, cSt + BATCH * HPW, h1w, KT1, wg, wid, lane, tid);
    grid_barrier(cnt, gen);

    // ---------------- Phase C: logits = h1_new @ out_w^T + out_b ----------------
    stage_h(h1w, aLds, tid);
    __syncthreads();
    if (wid < 4) {                                      // wave w -> m-tile w, n-tile = wg
      v8f acc = {};
      const v16h* bp   = ((const v16h*)bpo) + (long)wg * KTO * 32 + lane;
      const int   mrow = lane & 15;
      const int   klo  = (lane & 16) ? 8 : 0;
      for (int kt = 0; kt < KTO; ++kt) {
        v16h bfrag = bp[(long)kt * 32];
        const _Float16* ap = aLds + (wid * 16 + mrow) * ASTR + kt * 32 + klo;
        v8h lo = *(const v8h*)ap;
        v8h hi = *(const v8h*)(ap + 16);
        v16h af = __builtin_shufflevector(lo, hi, 0,1,2,3,4,5,6,7,8,9,10,11,12,13,14,15);
        acc = __builtin_amdgcn_wmma_f32_16x16x32_f16(false, af, false, bfrag,
                                                     (short)0, acc, false, false);
      }
      const int v    = wg * 16 + (lane & 15);
      const float ob = out_b[v];
      const int rbase = (lane & 16) ? 8 : 0;
#pragma unroll
      for (int r = 0; r < 8; ++r) {
        int b = wid * 16 + rbase + r;
        float val = acc[r] + ob;
        out[((long)b * SEQ + t) * VOCAB + v] = val;     // [B, T, 1, V]
        logits[b * VOCAB + v] = val;                    // feed next step's softmax
      }
    }
    grid_barrier(cnt, gen);
  }
}

extern "C" void kernel_launch(void* const* d_in, const int* in_sizes, int n_in,
                              void* d_out, int out_size, void* d_ws, size_t ws_size,
                              hipStream_t stream) {
  const float* z     = (const float*)d_in[0];
  const float* fc_w  = (const float*)d_in[1];
  const float* fc_b  = (const float*)d_in[2];
  const float* w_ih0 = (const float*)d_in[3];
  const float* w_hh0 = (const float*)d_in[4];
  const float* b_ih0 = (const float*)d_in[5];
  const float* b_hh0 = (const float*)d_in[6];
  const float* w_ih1 = (const float*)d_in[7];
  const float* w_hh1 = (const float*)d_in[8];
  const float* b_ih1 = (const float*)d_in[9];
  const float* b_hh1 = (const float*)d_in[10];
  const float* out_w = (const float*)d_in[11];
  const float* out_b = (const float*)d_in[12];

  char* ws = (char*)d_ws;
  unsigned*  bar    = (unsigned*)(ws + WS_BAR);
  float*     b0     = (float*)(ws + WS_B0);
  float*     b1     = (float*)(ws + WS_B1);
  _Float16*  h0buf  = (_Float16*)(ws + WS_H0);
  _Float16*  h1buf  = (_Float16*)(ws + WS_H1);
  float*     logits = (float*)(ws + WS_LOG);
  _Float16*  bp0    = (_Float16*)(ws + WS_BP0);
  _Float16*  bp1    = (_Float16*)(ws + WS_BP1);
  _Float16*  bpo    = (_Float16*)(ws + WS_BPO);

  const long t0 = 256L * KT0 * 512;   // f16 elements per pack
  const long t1 = 256L * KT1 * 512;
  const long to = 32L  * KTO * 512;
  pack_b_kernel<<<(int)((t0 + 255) / 256), 256, 0, stream>>>(w_ih0, w_hh0, 512, 1024, KT0, t0, bp0);
  pack_b_kernel<<<(int)((t1 + 255) / 256), 256, 0, stream>>>(w_ih1, w_hh1, 1024, 1024, KT1, t1, bp1);
  pack_b_kernel<<<(int)((to + 255) / 256), 256, 0, stream>>>(out_w, out_w, 1024, 0, KTO, to, bpo);
  init_kernel<<<256, 256, 0, stream>>>(z, fc_w, fc_b, b_ih0, b_hh0, b_ih1, b_hh1,
                                       h0buf, h1buf, b0, b1, bar);

  const size_t smem = (size_t)BATCH * ASTR * 2      // A buffer
                    + (size_t)BATCH * GSTR * 4      // gates
                    + (size_t)2 * BATCH * HPW * 4;  // cell state (2 layers)
  lstm_main<<<NWG, NTHREADS, smem, stream>>>(bp0, bp1, bpo, b0, b1, out_b,
                                             h0buf, h1buf, logits, (float*)d_out, bar);
}
